// CDQNModel_88321707475045
// MI455X (gfx1250) — matrix operand order repaired and checked by reading, not compiled, not run on hardware
//
#include <hip/hip_runtime.h>

// ---------------------------------------------------------------------------
// CDNA5 / gfx1250 CDQN forward pass.
// Core: zero-padded bf16 WMMA GEMM, async global->LDS staging (ASYNCcnt),
// double-buffered LDS, 128x128 block tile, 32x64 per wave (2x4 WMMA),
// v_wmma_f32_16x16x32_bf16.
// All GEMMs here have M % 128 == 0 and N % 128 == 0; K zero-padded to a
// multiple of 32 (exact), so the hot loop has NO bounds checks.
// ---------------------------------------------------------------------------

typedef __attribute__((ext_vector_type(16))) __bf16 bf16x16;
typedef __attribute__((ext_vector_type(8)))  float  f32x8;

__device__ __forceinline__ unsigned short f2bf_bits(float f) {
    unsigned u = __float_as_uint(f);
    return (unsigned short)((u + 0x7FFFu + ((u >> 16) & 1u)) >> 16);   // RNE
}
__device__ __forceinline__ float bf2f(unsigned short s) {
    return __uint_as_float(((unsigned)s) << 16);
}

// gfx1250 async copy: LDS[lds_off] = MEM[gaddr], 16 bytes, tracked on ASYNCcnt.
__device__ __forceinline__ void async_b128(unsigned lds_off, const void* gaddr) {
    asm volatile("global_load_async_to_lds_b128 %0, %1, off"
                 :: "v"(lds_off), "v"((unsigned long long)(uintptr_t)gaddr)
                 : "memory");
}
__device__ __forceinline__ void wait_async0() {
    asm volatile("s_wait_asynccnt 0" ::: "memory");
}

// ---------------------------------------------------------------------------
// out[M,N](bf16) = relu(A[M,lda](bf16, zero-padded) * WT^T + bias) [+ res]
// WT: weights pre-transposed/converted: WT[N][ldw] bf16, zero-padded.
// Block: 256 thr = 8 waves; tile 128(M) x 128(N); K-step 32.
// Wave grid 4(M) x 2(N); each wave: 32 rows x 64 cols = 2x4 WMMA tiles.
// ---------------------------------------------------------------------------
__global__ __launch_bounds__(256)
void k_gemm_bf16(const unsigned short* __restrict__ A, int lda,
                 const unsigned short* __restrict__ WT, int ldw,
                 const float*          __restrict__ bias,
                 const unsigned short* __restrict__ res,   // bf16 or null
                 unsigned short*       __restrict__ out,
                 int M, int N, int K32)                    // K32 % 32 == 0
{
    __shared__ unsigned short lA[2][128][40];   // [buf][m][k]
    __shared__ unsigned short lB[2][128][40];   // [buf][n][k] (transposed W)

    const int tid    = threadIdx.x;
    const int lane   = tid & 31;
    const int wid    = tid >> 5;
    const int waveM  = wid & 3;                 // 0..3 -> 32 rows each
    const int waveN  = wid >> 2;                // 0..1 -> 64 cols each
    const int blockM = blockIdx.y * 128;
    const int blockN = blockIdx.x * 128;

    // staging coordinates: each thread copies 32B of A and 32B of B per step
    const int sm  = tid >> 1;                   // 0..127
    const int skh = (tid & 1) << 4;             // 0 or 16
    const unsigned short* gA = A  + (size_t)(blockM + sm) * lda + skh;
    const unsigned short* gB = WT + (size_t)(blockN + sm) * ldw + skh;

    union Frag { bf16x16 v; unsigned int u[8]; };

    f32x8 acc[2][4];
#pragma unroll
    for (int i = 0; i < 2; ++i)
#pragma unroll
        for (int j = 0; j < 4; ++j) acc[i][j] = (f32x8)(0.0f);

    const int mL = lane & 15;
    const int hh = lane >> 4;

    // prologue: stage tile 0 into buffer 0
    async_b128((unsigned)(uintptr_t)&lA[0][sm][skh],     gA);
    async_b128((unsigned)(uintptr_t)&lA[0][sm][skh + 8], gA + 8);
    async_b128((unsigned)(uintptr_t)&lB[0][sm][skh],     gB);
    async_b128((unsigned)(uintptr_t)&lB[0][sm][skh + 8], gB + 8);

    int p = 0;
    for (int k0 = 0; k0 < K32; k0 += 32, p ^= 1) {
        wait_async0();          // own async stores to LDS landed
        __syncthreads();        // all waves' stores landed; prev reads done

        if (k0 + 32 < K32) {    // stage next tile into the other buffer
            const int kn = k0 + 32;
            async_b128((unsigned)(uintptr_t)&lA[p ^ 1][sm][skh],     gA + kn);
            async_b128((unsigned)(uintptr_t)&lA[p ^ 1][sm][skh + 8], gA + kn + 8);
            async_b128((unsigned)(uintptr_t)&lB[p ^ 1][sm][skh],     gB + kn);
            async_b128((unsigned)(uintptr_t)&lB[p ^ 1][sm][skh + 8], gB + kn + 8);
        }

        // fragment gathers per ISA 7.12.2 (contiguous 32-bit LDS reads)
        Frag fa[2], fb[4];
#pragma unroll
        for (int i = 0; i < 2; ++i) {
            const unsigned short* ap = &lA[p][waveM * 32 + i * 16 + mL][0];
#pragma unroll
            for (int q = 0; q < 4; ++q) {
                fa[i].u[q]     = *(const unsigned int*)(ap + 8 * hh + 2 * q);
                fa[i].u[4 + q] = *(const unsigned int*)(ap + 16 + 8 * hh + 2 * q);
            }
        }
#pragma unroll
        for (int j = 0; j < 4; ++j) {
            const unsigned short* bp = &lB[p][waveN * 64 + j * 16 + mL][0];
#pragma unroll
            for (int q = 0; q < 8; ++q)
                fb[j].u[q] = *(const unsigned int*)(bp + 8 * hh + 2 * q + (q >= 4 ? 8 : 0));
        }

#pragma unroll
        for (int i = 0; i < 2; ++i)
#pragma unroll
            for (int j = 0; j < 4; ++j)
                acc[i][j] = __builtin_amdgcn_wmma_f32_16x16x32_bf16(
                    false, fa[i].v, false, fb[j].v, (short)0, acc[i][j], false, false);
    }

    // epilogue: bias + relu (+ residual); M%128==0, N%128==0 -> no guards
    const int mb = blockM + waveM * 32;
    const int nb = blockN + waveN * 64;
#pragma unroll
    for (int i = 0; i < 2; ++i) {
#pragma unroll
        for (int j = 0; j < 4; ++j) {
            const int n  = nb + j * 16 + mL;
            const float bv = bias[n];
#pragma unroll
            for (int r = 0; r < 8; ++r) {
                const int m = mb + i * 16 + hh * 8 + r;
                float v = acc[i][j][r] + bv;
                v = v > 0.0f ? v : 0.0f;
                if (res) v += bf2f(res[(size_t)m * N + n]);
                out[(size_t)m * N + n] = f2bf_bits(v);
            }
        }
    }
}

// ---- weight prep: WT[n*Kp + k] = bf16(W[k*N + n]), zero-padded k >= K ------
__global__ void k_wt_cvt(const float* __restrict__ W, unsigned short* __restrict__ WT,
                         int K, int N, int Kp)
{
    int idx = blockIdx.x * blockDim.x + threadIdx.x;
    if (idx >= N * Kp) return;
    const int n = idx / Kp, k = idx % Kp;
    WT[idx] = (k < K) ? f2bf_bits(W[(size_t)k * N + n]) : (unsigned short)0;
}

// ---- activation pack: out[row*ldo + c] = bf16(in[row*K + c]) (pad 0) -------
__global__ void k_cvt_pad(const float* __restrict__ in, unsigned short* __restrict__ out,
                          int rows, int K, int ldo)
{
    int idx = blockIdx.x * blockDim.x + threadIdx.x;
    if (idx >= rows * ldo) return;
    const int r = idx / ldo, c = idx % ldo;
    out[idx] = (c < K) ? f2bf_bits(in[(size_t)r * K + c]) : (unsigned short)0;
}

// ---- layer_norm(scale=False) + beta: one 256-thread block per row ----------
__global__ __launch_bounds__(256)
void k_layernorm(const unsigned short* __restrict__ y, const float* __restrict__ beta,
                 unsigned short* __restrict__ out, int Wd)
{
    const int row = blockIdx.x;
    const unsigned short* yr = y + (size_t)row * Wd;
    float s = 0.f, s2 = 0.f;
    for (int c = threadIdx.x; c < Wd; c += 256) {
        float v = bf2f(yr[c]); s += v; s2 += v * v;
    }
    for (int o = 16; o; o >>= 1) { s += __shfl_xor(s, o, 32); s2 += __shfl_xor(s2, o, 32); }
    __shared__ float sh[2][8];
    const int wid = threadIdx.x >> 5, lane = threadIdx.x & 31;
    if (lane == 0) { sh[0][wid] = s; sh[1][wid] = s2; }
    __syncthreads();
    if (threadIdx.x == 0) {
        float a = 0.f, b = 0.f;
#pragma unroll
        for (int i = 0; i < 8; ++i) { a += sh[0][i]; b += sh[1][i]; }
        sh[0][0] = a; sh[1][0] = b;
    }
    __syncthreads();
    const float mu  = sh[0][0] / (float)Wd;
    const float var = sh[1][0] / (float)Wd - mu * mu;
    const float inv = rsqrtf(var + 1e-12f);
    unsigned short* orow = out + (size_t)row * Wd;
    for (int c = threadIdx.x; c < Wd; c += 256)
        orow[c] = f2bf_bits((bf2f(yr[c]) - mu) * inv + beta[c]);
}

__global__ void k_max_s2(const unsigned short* __restrict__ gf, unsigned short* __restrict__ outp,
                         int S2, int H, int total)
{
    int idx = blockIdx.x * blockDim.x + threadIdx.x;
    if (idx >= total) return;
    const int row = idx / H, h = idx % H;
    const unsigned short* p = gf + ((size_t)row * S2) * H + h;
    float m = -3.4e38f;
    for (int s = 0; s < S2; ++s) { float v = bf2f(p[(size_t)s * H]); m = v > m ? v : m; }
    outp[idx] = f2bf_bits(m);
}

// sf[b,s2,:] = concat(gfmax[b*S1, :H], bf16(joint[b,0,s2,:F])), pad to ld 1088
__global__ void k_build_sf(const unsigned short* __restrict__ gfmax, const float* __restrict__ joint,
                           unsigned short* __restrict__ sf, int Bn, int S1, int S2, int F, int H,
                           int ld)
{
    int idx = blockIdx.x * blockDim.x + threadIdx.x;
    if (idx >= Bn * S2 * ld) return;
    const int row = idx / ld, c = idx % ld;
    const int b = row / S2, s2 = row % S2;
    unsigned short v = 0;
    if (c < H)          v = gfmax[((size_t)b * S1) * H + c];
    else if (c < H + F) v = f2bf_bits(joint[(((size_t)b * S1) * S2 + s2) * F + (c - H)]);
    sf[idx] = v;
}

// dueling head with A:[N,1] reduces to q = x @ wV + bV ; one wave per row
__global__ __launch_bounds__(256)
void k_head(const unsigned short* __restrict__ x, const float* __restrict__ wV,
            const float* __restrict__ bV, float* __restrict__ q, int rows, int K)
{
    const int wid = threadIdx.x >> 5, lane = threadIdx.x & 31;
    const int row = blockIdx.x * 8 + wid;
    if (row >= rows) return;
    const unsigned short* xr = x + (size_t)row * K;
    float s = 0.f;
    for (int k = lane; k < K; k += 32) s += bf2f(xr[k]) * wV[k];
    for (int o = 16; o; o >>= 1) s += __shfl_xor(s, o, 32);
    if (lane == 0) q[row] = s + bV[0];
}

__global__ void k_final(const float* __restrict__ q_comb, const float* __restrict__ q_fine,
                        const int* __restrict__ comb_mask, const int* __restrict__ fine_mask,
                        float* __restrict__ out, int Bn, int S1, int S2)
{
    int idx = blockIdx.x * blockDim.x + threadIdx.x;
    if (idx >= Bn * S2) return;
    const int b = idx / S2, j = idx % S2;
    const bool cm = comb_mask[b] != 0;
    const float qc  = (j < S1 && cm) ? q_comb[b * S1 + j] : 0.f;
    const float qf  = (!cm) ? q_fine[idx] : 0.f;
    const float pad = (cm && j >= S1) ? -100000.0f : 0.f;
    const float fm  = (fine_mask[idx] != 0) ? 0.f : 100000.0f;
    out[idx] = qc + qf + pad - fm;
}

// ---------------------------------------------------------------------------
// Host orchestration
// ---------------------------------------------------------------------------
extern "C" void kernel_launch(void* const* d_in, const int* in_sizes, int n_in,
                              void* d_out, int out_size, void* d_ws, size_t ws_size,
                              hipStream_t stream)
{
    (void)in_sizes; (void)n_in; (void)out_size; (void)ws_size;

    const int B = 16, S1 = 32, S2 = 64, F = 60, H = 1024;
    const int ROWS_G = B * S1 * S2;          // 32768
    const int ROWS_C = B * S1;               // 512
    const int ROWS_F = B * S2;               // 1024

    const float* joint     = (const float*)d_in[0];
    const int*   comb_mask = (const int*)d_in[1];
    const int*   fine_mask = (const int*)d_in[2];

    // leaf order: comb(27), comb_head(bA,bV,wA,wV), fine(27), fine_head(4), global(27)
    // per block (9): beta, bs, w1, b1, w2, b2, w3, b3, ws
    const int base_comb = 3, base_combH = 30, base_fine = 34, base_fineH = 61, base_glob = 65;
    auto P = [&](int i) { return (const float*)d_in[i]; };

    char* ws = (char*)d_ws;
    size_t off = 0;
    auto alloc = [&](size_t bytes) -> void* {
        void* p = ws + off; off += (bytes + 255) & ~(size_t)255; return p;
    };
    auto kp32 = [](int K) { return (K + 31) & ~31; };

    unsigned short* X0 = (unsigned short*)alloc((size_t)ROWS_G * 64 * 2);     // ld 64
    unsigned short* SA = (unsigned short*)alloc((size_t)ROWS_G * 1024 * 2);
    unsigned short* SB = (unsigned short*)alloc((size_t)ROWS_G * 1024 * 2);
    unsigned short* SC = (unsigned short*)alloc((size_t)ROWS_G * 1024 * 2);
    unsigned short* SD = (unsigned short*)alloc((size_t)ROWS_G * 1024 * 2);
    unsigned short* GM = (unsigned short*)alloc((size_t)ROWS_C * 1024 * 2);
    unsigned short* SF = (unsigned short*)alloc((size_t)ROWS_F * 1088 * 2);   // ld 1088
    unsigned short* cA = (unsigned short*)alloc((size_t)ROWS_C * 512 * 2);
    unsigned short* cB = (unsigned short*)alloc((size_t)ROWS_C * 512 * 2);
    unsigned short* cC = (unsigned short*)alloc((size_t)ROWS_C * 512 * 2);
    unsigned short* cD = (unsigned short*)alloc((size_t)ROWS_C * 512 * 2);
    unsigned short* fA = (unsigned short*)alloc((size_t)ROWS_F * 512 * 2);
    unsigned short* fB = (unsigned short*)alloc((size_t)ROWS_F * 512 * 2);
    unsigned short* fC = (unsigned short*)alloc((size_t)ROWS_F * 512 * 2);
    unsigned short* fD = (unsigned short*)alloc((size_t)ROWS_F * 512 * 2);
    float* q_comb = (float*)alloc((size_t)ROWS_C * 4);
    float* q_fine = (float*)alloc((size_t)ROWS_F * 4);

    // gemm: prep transposed bf16 weights, then WMMA GEMM
    auto gemm = [&](const unsigned short* Ap, int lda, const float* Wp, const float* bp,
                    const unsigned short* rp, unsigned short* op, int M, int N, int K) {
        const int Kp = kp32(K);
        unsigned short* WT = (unsigned short*)alloc((size_t)N * Kp * 2);
        k_wt_cvt<<<((size_t)N * Kp + 255) / 256, 256, 0, stream>>>(Wp, WT, K, N, Kp);
        dim3 g(N / 128, M / 128);
        k_gemm_bf16<<<g, 256, 0, stream>>>(Ap, lda, WT, Kp, bp, rp, op, M, N, Kp);
    };
    auto ln = [&](const unsigned short* y, const float* beta, unsigned short* o, int M, int Wd) {
        k_layernorm<<<M, 256, 0, stream>>>(y, beta, o, Wd);
    };
    // residual block: relu-stack(x) + relu(x Ws + bs), then LN + beta
    auto block = [&](const unsigned short* x, int lda, int M, int d, int u, int pb,
                     unsigned short* sA, unsigned short* sB, unsigned short* sC,
                     unsigned short* oLN) {
        gemm(x,  lda, P(pb + 2), P(pb + 3), nullptr, sA, M, u, d);   // t1
        gemm(sA, u,   P(pb + 4), P(pb + 5), nullptr, sB, M, u, u);   // t2
        gemm(x,  lda, P(pb + 8), P(pb + 1), nullptr, sC, M, u, d);   // shortcut
        gemm(sB, u,   P(pb + 6), P(pb + 7), sC,      sA, M, u, u);   // t3 + sc
        ln(sA, P(pb + 0), oLN, M, u);
    };

    // ---- global tower over all 32768 rows ----
    k_cvt_pad<<<((size_t)ROWS_G * 64 + 255) / 256, 256, 0, stream>>>(joint, X0, ROWS_G, F, 64);
    block(X0, 64,   ROWS_G,  60,  256, base_glob + 0 * 9, SA, SB, SC, SD);
    block(SD, 256,  ROWS_G, 256,  512, base_glob + 1 * 9, SA, SB, SC, SD);
    block(SD, 512,  ROWS_G, 512, 1024, base_glob + 2 * 9, SA, SB, SC, SD);   // gf = SD
    k_max_s2<<<(ROWS_C * H + 255) / 256, 256, 0, stream>>>(SD, GM, S2, H, ROWS_C * H);

    // ---- comb branch: rows = 512 ----
    block(GM, 1024, ROWS_C, 1024, 512, base_comb + 0 * 9, cA, cB, cC, cD);
    block(cD, 512,  ROWS_C,  512, 256, base_comb + 1 * 9, cA, cB, cC, cD);
    block(cD, 256,  ROWS_C,  256, 128, base_comb + 2 * 9, cA, cB, cC, cD);
    k_head<<<(ROWS_C + 7) / 8, 256, 0, stream>>>(cD, P(base_combH + 3), P(base_combH + 1),
                                                 q_comb, ROWS_C, 128);

    // ---- fine branch: rows = 1024, K = 1084 (ld 1088) ----
    k_build_sf<<<((size_t)ROWS_F * 1088 + 255) / 256, 256, 0, stream>>>(GM, joint, SF,
                                                                        B, S1, S2, F, H, 1088);
    block(SF, 1088, ROWS_F, 1084, 512, base_fine + 0 * 9, fA, fB, fC, fD);
    block(fD, 512,  ROWS_F,  512, 256, base_fine + 1 * 9, fA, fB, fC, fD);
    block(fD, 256,  ROWS_F,  256, 128, base_fine + 2 * 9, fA, fB, fC, fD);
    k_head<<<(ROWS_F + 7) / 8, 256, 0, stream>>>(fD, P(base_fineH + 3), P(base_fineH + 1),
                                                 q_fine, ROWS_F, 128);

    // ---- combine ----
    k_final<<<(ROWS_F + 255) / 256, 256, 0, stream>>>(q_comb, q_fine, comb_mask, fine_mask,
                                                      (float*)d_out, B, S1, S2);
}
